// GraphTransformerLayer_69982197121736
// MI455X (gfx1250) — matrix operand order repaired and checked by reading, not compiled, not run on hardware
//
#include <hip/hip_runtime.h>
#include <hip/hip_bf16.h>

// ---------------------------------------------------------------------------
// Problem constants (from reference)
// ---------------------------------------------------------------------------
#define N_NODES  20000
#define N_EDGES  320000
#define N_FEDGES 500000
#define DD       256
#define HH       8
#define HDIM     32
#define NGRAPH   100
#define RSCALE   0.17677669529663687f   // 1/sqrt(32)

typedef __attribute__((ext_vector_type(16))) __bf16 v16bf;
typedef __attribute__((ext_vector_type(8)))  __bf16 v8bf;
typedef __attribute__((ext_vector_type(8)))  float  v8f;

// ---------------------------------------------------------------------------
// Weight prep: f32 [K,Nc] -> bf16 transposed [Nc,K]  (one-time, tiny)
// ---------------------------------------------------------------------------
__global__ __launch_bounds__(256)
void cvt_transpose_bf16(const float* __restrict__ in, __bf16* __restrict__ out,
                        int K, int Nc)
{
    int i = blockIdx.x * 256 + threadIdx.x;   // i = n*K + k
    if (i >= K * Nc) return;
    int n = i / K, k = i % K;
    out[i] = (__bf16)in[(size_t)k * Nc + n];
}

// ---------------------------------------------------------------------------
// WMMA GEMM:  out[M,Nc] = A[M,K] @ B[K,Nc] + bias (+resid) (+relu)
//   Bt is the pre-transposed bf16 weight: Bt[n*K + k].
//   Tile: 128x64 per workgroup, BK=32; 8 wave32 waves, wave w owns rows
//   [16w,16w+16) x all 64 cols -> 4 accumulators, 4 WMMAs per K-step.
//   All 4 B-fragments are preloaded so the WMMAs issue back-to-back.
// ---------------------------------------------------------------------------
#define BM 128
#define BN 64
#define BK 32
#define APAD 40   // LDS row stride (elems): 80B, multiple of 16B
#define BPAD 40

template<bool AF32, bool OUTBF, bool HASRES, bool RELU>
__global__ __launch_bounds__(256)
void gemm_wmma_kernel(const void* __restrict__ Aptr, const __bf16* __restrict__ Bt,
                      const float* __restrict__ bias, const float* __restrict__ resid,
                      void* __restrict__ outP, int M, int Nc, int K)
{
    __shared__ __align__(16) __bf16 As[BM * APAD];
    __shared__ __align__(16) __bf16 Bs[BN * BPAD];   // Bs[n*BPAD + k]

    const int tid  = threadIdx.x;
    const int lane = tid & 31;
    const int wid  = tid >> 5;                 // row strip 0..7
    const int rowBase = blockIdx.y * BM;
    const int colBase = blockIdx.x * BN;
    const int half = lane >> 4;
    const int mn   = lane & 15;

    v8f acc[4] = {{}, {}, {}, {}};

    // A stage: 128 rows x 32 cols, 16 elems / thread
    const int aRow = tid >> 1;                 // 0..127
    const int aCol = (tid & 1) << 4;           // 0 or 16
    const int gr   = rowBase + aRow;
    const bool aInRange = (gr < M);
    // B stage: 64 n x 32 k, 8 elems / thread (contiguous in Bt)
    const int bN = tid >> 2;                   // 0..63
    const int bK = (tid & 3) << 3;             // 0,8,16,24

    // Zero-fill out-of-range A rows ONCE (never overwritten below; the first
    // K-step's barrier makes this visible before any fragment load).
    if (!aInRange) {
        __bf16* dst = &As[aRow * APAD + aCol];
#pragma unroll
        for (int j = 0; j < 16; ++j) dst[j] = (__bf16)0.0f;
    }

    for (int kt = 0; kt < K; kt += BK) {
        // ---- stage A (predicated, no else path) ----
        if (aInRange) {
            __bf16* dst = &As[aRow * APAD + aCol];
            if (AF32) {
                const float* ap = (const float*)Aptr + (size_t)gr * K + kt + aCol;
                float4 t0 = ((const float4*)ap)[0];
                float4 t1 = ((const float4*)ap)[1];
                float4 t2 = ((const float4*)ap)[2];
                float4 t3 = ((const float4*)ap)[3];
                if (kt + BK < K) __builtin_prefetch(ap + BK, 0, 3);
                dst[0]  = (__bf16)t0.x; dst[1]  = (__bf16)t0.y;
                dst[2]  = (__bf16)t0.z; dst[3]  = (__bf16)t0.w;
                dst[4]  = (__bf16)t1.x; dst[5]  = (__bf16)t1.y;
                dst[6]  = (__bf16)t1.z; dst[7]  = (__bf16)t1.w;
                dst[8]  = (__bf16)t2.x; dst[9]  = (__bf16)t2.y;
                dst[10] = (__bf16)t2.z; dst[11] = (__bf16)t2.w;
                dst[12] = (__bf16)t3.x; dst[13] = (__bf16)t3.y;
                dst[14] = (__bf16)t3.z; dst[15] = (__bf16)t3.w;
            } else {
                const __bf16* ap = (const __bf16*)Aptr + (size_t)gr * K + kt + aCol;
                v8bf a0 = ((const v8bf*)ap)[0];
                v8bf a1 = ((const v8bf*)ap)[1];
                if (kt + BK < K) __builtin_prefetch(ap + BK, 0, 3);
                *(v8bf*)(dst)     = a0;
                *(v8bf*)(dst + 8) = a1;
            }
        }
        // ---- stage B (pre-transposed weights: contiguous load + 16B LDS store) ----
        {
            v8bf bv = *(const v8bf*)(Bt + (size_t)(colBase + bN) * K + kt + bK);
            *(v8bf*)(&Bs[bN * BPAD + bK]) = bv;
        }
        __syncthreads();

        // ---- preload A fragment + ALL 4 B fragments, then 4 back-to-back WMMAs
        const __bf16* arow = &As[(wid * 16 + mn) * APAD];
        v8bf aLo = *(const v8bf*)(arow + half * 8);        // K = half*8 + 0..7
        v8bf aHi = *(const v8bf*)(arow + 16 + half * 8);   // K = 16 + half*8 + 0..7
        v16bf afrag = __builtin_shufflevector(aLo, aHi,
                        0,1,2,3,4,5,6,7,8,9,10,11,12,13,14,15);

        v8bf bl0, bh0, bl1, bh1, bl2, bh2, bl3, bh3;
        {
            const __bf16* b0 = &Bs[(0 * 16 + mn) * BPAD + half * 8];
            const __bf16* b1 = &Bs[(1 * 16 + mn) * BPAD + half * 8];
            const __bf16* b2 = &Bs[(2 * 16 + mn) * BPAD + half * 8];
            const __bf16* b3 = &Bs[(3 * 16 + mn) * BPAD + half * 8];
            bl0 = *(const v8bf*)(b0); bh0 = *(const v8bf*)(b0 + 16);
            bl1 = *(const v8bf*)(b1); bh1 = *(const v8bf*)(b1 + 16);
            bl2 = *(const v8bf*)(b2); bh2 = *(const v8bf*)(b2 + 16);
            bl3 = *(const v8bf*)(b3); bh3 = *(const v8bf*)(b3 + 16);
        }
        v16bf bf0 = __builtin_shufflevector(bl0, bh0, 0,1,2,3,4,5,6,7,8,9,10,11,12,13,14,15);
        v16bf bf1 = __builtin_shufflevector(bl1, bh1, 0,1,2,3,4,5,6,7,8,9,10,11,12,13,14,15);
        v16bf bf2 = __builtin_shufflevector(bl2, bh2, 0,1,2,3,4,5,6,7,8,9,10,11,12,13,14,15);
        v16bf bf3 = __builtin_shufflevector(bl3, bh3, 0,1,2,3,4,5,6,7,8,9,10,11,12,13,14,15);

        acc[0] = __builtin_amdgcn_wmma_f32_16x16x32_bf16(false, afrag, false, bf0,
                                                         (short)0, acc[0], false, false);
        acc[1] = __builtin_amdgcn_wmma_f32_16x16x32_bf16(false, afrag, false, bf1,
                                                         (short)0, acc[1], false, false);
        acc[2] = __builtin_amdgcn_wmma_f32_16x16x32_bf16(false, afrag, false, bf2,
                                                         (short)0, acc[2], false, false);
        acc[3] = __builtin_amdgcn_wmma_f32_16x16x32_bf16(false, afrag, false, bf3,
                                                         (short)0, acc[3], false, false);
        __syncthreads();
    }

    // ---- epilogue ----
#pragma unroll
    for (int c = 0; c < 4; ++c) {
        const int col = colBase + c * 16 + mn;
        const float bcol = bias[col];
#pragma unroll
        for (int i = 0; i < 8; ++i) {
            int r = rowBase + wid * 16 + half * 8 + i;   // C/D layout: M = i + half*8
            if (r < M) {
                float v = acc[c][i] + bcol;
                if (HASRES) v += resid[(size_t)r * Nc + col];
                if (RELU)   v = fmaxf(v, 0.0f);
                if (OUTBF)  ((__bf16*)outP)[(size_t)r * Nc + col] = (__bf16)v;
                else        ((float*)outP)[(size_t)r * Nc + col] = v;
            }
        }
    }
}

// ---------------------------------------------------------------------------
// Sparse-graph edge kernel: wave per (edge, head); lane = head-dim (HD==32).
// ---------------------------------------------------------------------------
__global__ __launch_bounds__(256)
void edge_sparse_kernel(const float* __restrict__ Qh, const float* __restrict__ Kh,
                        const float* __restrict__ Vh, float* __restrict__ projE,
                        const int* __restrict__ src, const int* __restrict__ dst,
                        float* __restrict__ wV, float* __restrict__ z)
{
    int gw = blockIdx.x * 8 + (threadIdx.x >> 5);
    if (gw >= N_EDGES * HH) return;
    int lane = threadIdx.x & 31;
    int e = gw >> 3;
    int h = gw & 7;
    int s_ = src[e], d_ = dst[e];
    int col = h * HDIM + lane;

    float kv = Kh[(size_t)s_ * DD + col];
    float qv = Qh[(size_t)d_ * DD + col];
    float pv = projE[(size_t)e * DD + col];
    float score = kv * qv * RSCALE * pv;
    projE[(size_t)e * DD + col] = score;          // e_out (in place)

    float sum = score;
#pragma unroll
    for (int off = 16; off > 0; off >>= 1) sum += __shfl_xor(sum, off, 32);
    float sexp = expf(fminf(fmaxf(sum, -5.0f), 5.0f));

    float vv = Vh[(size_t)s_ * DD + col];
    atomicAdd(&wV[(size_t)d_ * DD + col], vv * sexp);
    if (lane == 0) atomicAdd(&z[d_ * HH + h], sexp);
}

// ---------------------------------------------------------------------------
// Full-graph edge kernel: wave per (fedge, head).
// ---------------------------------------------------------------------------
__global__ __launch_bounds__(256)
void edge_full_kernel(const float* __restrict__ Qh, const float* __restrict__ Kh,
                      const float* __restrict__ Vh, const float* __restrict__ rel,
                      const float* __restrict__ adj2,
                      const int* __restrict__ fsrc, const int* __restrict__ fdst,
                      float* __restrict__ wVf, float* __restrict__ zf)
{
    int gw = blockIdx.x * 8 + (threadIdx.x >> 5);
    if (gw >= N_FEDGES * HH) return;
    int lane = threadIdx.x & 31;
    int fe = gw >> 3;
    int h  = gw & 7;
    int s_ = fsrc[fe], d_ = fdst[fe];
    int col = h * HDIM + lane;

    float dot = Kh[(size_t)s_ * DD + col] * Qh[(size_t)d_ * DD + col];
#pragma unroll
    for (int off = 16; off > 0; off >>= 1) dot += __shfl_xor(dot, off, 32);
    float sc = dot * RSCALE;
    float sf = expf(fminf(fmaxf(sc + rel[fe * HH + h], -5.0f), 5.0f) * adj2[fe]);

    float vv = Vh[(size_t)s_ * DD + col];
    atomicAdd(&wVf[(size_t)d_ * DD + col], vv * sf);
    if (lane == 0) atomicAdd(&zf[d_ * HH + h], sf);
}

// ---------------------------------------------------------------------------
// h_attn = wV/(z+1e-6) + wVf/(zf+1e-6)   (in place into wV)
// ---------------------------------------------------------------------------
__global__ __launch_bounds__(256)
void attn_combine_kernel(float* __restrict__ wV, const float* __restrict__ z,
                         const float* __restrict__ wVf, const float* __restrict__ zf)
{
    int i = blockIdx.x * 256 + threadIdx.x;
    if (i >= N_NODES * DD) return;
    int row = i >> 8;
    int hh  = (i & 255) >> 5;
    float a = wV[i]  / (z[row * HH + hh]  + 1e-6f);
    float b = wVf[i] / (zf[row * HH + hh] + 1e-6f);
    wV[i] = a + b;
}

// ---------------------------------------------------------------------------
// LayerNorm over D=256, wave per row, in place.
// ---------------------------------------------------------------------------
__global__ __launch_bounds__(256)
void layernorm_kernel(float* __restrict__ x, const float* __restrict__ g,
                      const float* __restrict__ b, int rows)
{
    int row = blockIdx.x * 8 + (threadIdx.x >> 5);
    if (row >= rows) return;
    int lane = threadIdx.x & 31;
    float* xr = x + (size_t)row * DD + lane * 8;
    float4 a0 = *(float4*)(xr);
    float4 a1 = *(float4*)(xr + 4);
    float v[8] = {a0.x,a0.y,a0.z,a0.w,a1.x,a1.y,a1.z,a1.w};

    float s = 0.0f;
#pragma unroll
    for (int j = 0; j < 8; ++j) s += v[j];
#pragma unroll
    for (int off = 16; off > 0; off >>= 1) s += __shfl_xor(s, off, 32);
    float mu = s * (1.0f / DD);

    float q = 0.0f;
#pragma unroll
    for (int j = 0; j < 8; ++j) { float d = v[j] - mu; q += d * d; }
#pragma unroll
    for (int off = 16; off > 0; off >>= 1) q += __shfl_xor(q, off, 32);
    float rstd = rsqrtf(q * (1.0f / DD) + 1e-5f);

#pragma unroll
    for (int j = 0; j < 8; ++j) {
        int c = lane * 8 + j;
        xr[j] = (v[j] - mu) * rstd * g[c] + b[c];
    }
}

// ---------------------------------------------------------------------------
// GraphNorm (NG=100): count, sum, sumsq(sub), finalize (in place).
// ---------------------------------------------------------------------------
__global__ __launch_bounds__(256)
void gn_count_kernel(const int* __restrict__ gid, float* __restrict__ cnt)
{
    int i = blockIdx.x * 256 + threadIdx.x;
    if (i < N_NODES) atomicAdd(&cnt[gid[i]], 1.0f);
}

__global__ __launch_bounds__(256)
void gn_sum_kernel(const float* __restrict__ x, const int* __restrict__ gid,
                   float* __restrict__ sum)
{
    int i = blockIdx.x * 256 + threadIdx.x;
    if (i >= N_NODES * DD) return;
    int row = i >> 8, c = i & 255;
    atomicAdd(&sum[gid[row] * DD + c], x[i]);
}

__global__ __launch_bounds__(256)
void gn_sumsq_kernel(const float* __restrict__ x, const int* __restrict__ gid,
                     const float* __restrict__ sum, const float* __restrict__ cnt,
                     const float* __restrict__ alpha, float* __restrict__ ss)
{
    int i = blockIdx.x * 256 + threadIdx.x;
    if (i >= N_NODES * DD) return;
    int row = i >> 8, c = i & 255;
    int g = gid[row];
    float cg = fmaxf(cnt[g], 1.0f);
    float mean = sum[g * DD + c] / cg;
    float sub = x[i] - alpha[c] * mean;
    atomicAdd(&ss[g * DD + c], sub * sub);
}

__global__ __launch_bounds__(256)
void gn_final_kernel(float* __restrict__ x, const int* __restrict__ gid,
                     const float* __restrict__ sum, const float* __restrict__ ss,
                     const float* __restrict__ cnt, const float* __restrict__ alpha,
                     const float* __restrict__ gamma, const float* __restrict__ beta)
{
    int i = blockIdx.x * 256 + threadIdx.x;
    if (i >= N_NODES * DD) return;
    int row = i >> 8, c = i & 255;
    int g = gid[row];
    float cg = fmaxf(cnt[g], 1.0f);
    float mean = sum[g * DD + c] / cg;
    float sub = x[i] - alpha[c] * mean;
    float var = ss[g * DD + c] / cg;
    x[i] = gamma[c] * sub * rsqrtf(var + 1e-6f) + beta[c];
}

// ---------------------------------------------------------------------------
// Host launcher
// ---------------------------------------------------------------------------
extern "C" void kernel_launch(void* const* d_in, const int* in_sizes, int n_in,
                              void* d_out, int out_size, void* d_ws, size_t ws_size,
                              hipStream_t stream)
{
    (void)in_sizes; (void)n_in; (void)out_size; (void)ws_size;

    const float* hx   = (const float*)d_in[0];
    const float* ex   = (const float*)d_in[1];
    const int*   src  = (const int*)d_in[2];
    const int*   dst  = (const int*)d_in[3];
    const int*   fsrc = (const int*)d_in[4];
    const int*   fdst = (const int*)d_in[5];
    const float* adj2 = (const float*)d_in[6];
    const float* rel  = (const float*)d_in[7];
    const int*   gid  = (const int*)d_in[8];
    const float* Qw   = (const float*)d_in[9];
    const float* Kw   = (const float*)d_in[10];
    const float* Vw   = (const float*)d_in[11];
    const float* Pw   = (const float*)d_in[12];
    const float* Ohw  = (const float*)d_in[13];
    const float* Oew  = (const float*)d_in[14];
    const float* Qb   = (const float*)d_in[15];
    const float* Kb   = (const float*)d_in[16];
    const float* Vb   = (const float*)d_in[17];
    const float* Pb   = (const float*)d_in[18];
    const float* Ohb  = (const float*)d_in[19];
    const float* Oeb  = (const float*)d_in[20];
    const float* f1hw = (const float*)d_in[21];
    const float* f1hb = (const float*)d_in[22];
    const float* f2hw = (const float*)d_in[23];
    const float* f2hb = (const float*)d_in[24];
    const float* f1ew = (const float*)d_in[25];
    const float* f1eb = (const float*)d_in[26];
    const float* f2ew = (const float*)d_in[27];
    const float* f2eb = (const float*)d_in[28];
    const float* gn1_alpha = (const float*)d_in[29];
    const float* gn1_gamma = (const float*)d_in[30];
    const float* gn2_alpha = (const float*)d_in[31];
    const float* gn2_gamma = (const float*)d_in[32];
    const float* ln1e_g    = (const float*)d_in[33];
    const float* ln2e_g    = (const float*)d_in[34];
    const float* gn1_beta  = (const float*)d_in[35];
    const float* gn2_beta  = (const float*)d_in[36];
    const float* ln1e_b    = (const float*)d_in[37];
    const float* ln2e_b    = (const float*)d_in[38];

    float* outH = (float*)d_out;                       // [N, D]
    float* outE = outH + (size_t)N_NODES * DD;         // [E, D]

    // ---- workspace carve (256B aligned) ----
    char*  base = (char*)d_ws;
    size_t off  = 0;
    auto alloc = [&](size_t bytes) -> void* {
        off = (off + 255) & ~(size_t)255;
        void* p = base + off;
        off += bytes;
        return p;
    };
    __bf16* wQ   = (__bf16*)alloc((size_t)DD * DD * 2);
    __bf16* wK   = (__bf16*)alloc((size_t)DD * DD * 2);
    __bf16* wVw  = (__bf16*)alloc((size_t)DD * DD * 2);
    __bf16* wP   = (__bf16*)alloc((size_t)DD * DD * 2);
    __bf16* wOh  = (__bf16*)alloc((size_t)DD * DD * 2);
    __bf16* wOe  = (__bf16*)alloc((size_t)DD * DD * 2);
    __bf16* wF1h = (__bf16*)alloc((size_t)DD * 2 * DD * 2);
    __bf16* wF2h = (__bf16*)alloc((size_t)DD * 2 * DD * 2);
    __bf16* wF1e = (__bf16*)alloc((size_t)DD * 2 * DD * 2);
    __bf16* wF2e = (__bf16*)alloc((size_t)DD * 2 * DD * 2);
    float* Qh    = (float*)alloc((size_t)N_NODES * DD * 4);
    float* Kh    = (float*)alloc((size_t)N_NODES * DD * 4);
    float* Vh    = (float*)alloc((size_t)N_NODES * DD * 4);
    float* projE = (float*)alloc((size_t)N_EDGES * DD * 4);   // becomes e_out
    float* wV    = (float*)alloc((size_t)N_NODES * DD * 4);   // becomes h_attn
    float* zb    = (float*)alloc((size_t)N_NODES * HH * 4);
    float* wVf   = (float*)alloc((size_t)N_NODES * DD * 4);
    float* zf    = (float*)alloc((size_t)N_NODES * HH * 4);
    float* gnSum = (float*)alloc((size_t)NGRAPH * DD * 4);
    float* gnSS  = (float*)alloc((size_t)NGRAPH * DD * 4);
    float* gnCnt = (float*)alloc((size_t)NGRAPH * 4);
    __bf16* hidH = (__bf16*)alloc((size_t)N_NODES * 2 * DD * 2);
    __bf16* hidE = (__bf16*)alloc((size_t)N_EDGES * 2 * DD * 2);

    // transpose+convert weights: in [K,Nc] -> out [Nc,K]
    auto cvtT = [&](const float* in, __bf16* out, int K, int Nc) {
        int n = K * Nc;
        cvt_transpose_bf16<<<(n + 255) / 256, 256, 0, stream>>>(in, out, K, Nc);
    };

    cvtT(Qw,  wQ,  DD, DD);  cvtT(Kw,  wK,  DD, DD);  cvtT(Vw, wVw, DD, DD);
    cvtT(Pw,  wP,  DD, DD);  cvtT(Ohw, wOh, DD, DD);  cvtT(Oew, wOe, DD, DD);
    cvtT(f1hw, wF1h, DD, 2 * DD);  cvtT(f2hw, wF2h, 2 * DD, DD);
    cvtT(f1ew, wF1e, DD, 2 * DD);  cvtT(f2ew, wF2e, 2 * DD, DD);

    // zero atomic accumulators
    hipMemsetAsync(wV,  0, (size_t)N_NODES * DD * 4, stream);
    hipMemsetAsync(zb,  0, (size_t)N_NODES * HH * 4, stream);
    hipMemsetAsync(wVf, 0, (size_t)N_NODES * DD * 4, stream);
    hipMemsetAsync(zf,  0, (size_t)N_NODES * HH * 4, stream);

    auto grid = [](int M, int Nc) { return dim3(Nc / BN, (M + BM - 1) / BM); };

    // input projections (A=f32, out=f32)
    gemm_wmma_kernel<true,false,false,false><<<grid(N_NODES, DD), 256, 0, stream>>>(
        hx, wQ, Qb, nullptr, Qh, N_NODES, DD, DD);
    gemm_wmma_kernel<true,false,false,false><<<grid(N_NODES, DD), 256, 0, stream>>>(
        hx, wK, Kb, nullptr, Kh, N_NODES, DD, DD);
    gemm_wmma_kernel<true,false,false,false><<<grid(N_NODES, DD), 256, 0, stream>>>(
        hx, wVw, Vb, nullptr, Vh, N_NODES, DD, DD);
    gemm_wmma_kernel<true,false,false,false><<<grid(N_EDGES, DD), 256, 0, stream>>>(
        ex, wP, Pb, nullptr, projE, N_EDGES, DD, DD);

    // sparse-graph attention (e_out in place; atomics into wV/z)
    edge_sparse_kernel<<<(N_EDGES * HH) / 8, 256, 0, stream>>>(Qh, Kh, Vh, projE,
                                                               src, dst, wV, zb);
    // full-graph attention
    edge_full_kernel<<<(N_FEDGES * HH) / 8, 256, 0, stream>>>(Qh, Kh, Vh, rel, adj2,
                                                              fsrc, fdst, wVf, zf);
    // combine into h_attn (in place into wV)
    attn_combine_kernel<<<(N_NODES * DD) / 256, 256, 0, stream>>>(wV, zb, wVf, zf);

    // output projections with residual (A=f32, resid, out=f32)
    gemm_wmma_kernel<true,false,true,false><<<grid(N_NODES, DD), 256, 0, stream>>>(
        wV, wOh, Ohb, hx, outH, N_NODES, DD, DD);
    gemm_wmma_kernel<true,false,true,false><<<grid(N_EDGES, DD), 256, 0, stream>>>(
        projE, wOe, Oeb, ex, outE, N_EDGES, DD, DD);

    // GraphNorm 1 on outH (in place)
    hipMemsetAsync(gnSum, 0, (size_t)NGRAPH * DD * 4, stream);
    hipMemsetAsync(gnSS,  0, (size_t)NGRAPH * DD * 4, stream);
    hipMemsetAsync(gnCnt, 0, (size_t)NGRAPH * 4, stream);
    gn_count_kernel<<<(N_NODES + 255) / 256, 256, 0, stream>>>(gid, gnCnt);
    gn_sum_kernel<<<(N_NODES * DD) / 256, 256, 0, stream>>>(outH, gid, gnSum);
    gn_sumsq_kernel<<<(N_NODES * DD) / 256, 256, 0, stream>>>(outH, gid, gnSum, gnCnt,
                                                              gn1_alpha, gnSS);
    gn_final_kernel<<<(N_NODES * DD) / 256, 256, 0, stream>>>(outH, gid, gnSum, gnSS,
                                                              gnCnt, gn1_alpha,
                                                              gn1_gamma, gn1_beta);
    // LayerNorm 1 on outE (in place)
    layernorm_kernel<<<(N_EDGES + 7) / 8, 256, 0, stream>>>(outE, ln1e_g, ln1e_b, N_EDGES);

    // node FFN: hidden = relu(outH @ f1hw + b) -> bf16; second GEMM in-place residual
    gemm_wmma_kernel<true,true,false,true><<<grid(N_NODES, 2 * DD), 256, 0, stream>>>(
        outH, wF1h, f1hb, nullptr, hidH, N_NODES, 2 * DD, DD);
    gemm_wmma_kernel<false,false,true,false><<<grid(N_NODES, DD), 256, 0, stream>>>(
        hidH, wF2h, f2hb, outH, outH, N_NODES, DD, 2 * DD);

    // GraphNorm 2 on outH (in place)
    hipMemsetAsync(gnSum, 0, (size_t)NGRAPH * DD * 4, stream);
    hipMemsetAsync(gnSS,  0, (size_t)NGRAPH * DD * 4, stream);
    gn_sum_kernel<<<(N_NODES * DD) / 256, 256, 0, stream>>>(outH, gid, gnSum);
    gn_sumsq_kernel<<<(N_NODES * DD) / 256, 256, 0, stream>>>(outH, gid, gnSum, gnCnt,
                                                              gn2_alpha, gnSS);
    gn_final_kernel<<<(N_NODES * DD) / 256, 256, 0, stream>>>(outH, gid, gnSum, gnSS,
                                                              gnCnt, gn2_alpha,
                                                              gn2_gamma, gn2_beta);

    // edge FFN
    gemm_wmma_kernel<true,true,false,true><<<grid(N_EDGES, 2 * DD), 256, 0, stream>>>(
        outE, wF1e, f1eb, nullptr, hidE, N_EDGES, 2 * DD, DD);
    gemm_wmma_kernel<false,false,true,false><<<grid(N_EDGES, DD), 256, 0, stream>>>(
        hidE, wF2e, f2eb, outE, outE, N_EDGES, DD, 2 * DD);

    // LayerNorm 2 on outE (in place)
    layernorm_kernel<<<(N_EDGES + 7) / 8, 256, 0, stream>>>(outE, ln2e_g, ln2e_b, N_EDGES);
}
